// SparseLayer_82377472737543
// MI455X (gfx1250) — matrix operand order repaired and checked by reading, not compiled, not run on hardware
//
#include <hip/hip_runtime.h>

// out[o,b] = sum_i W[i,o] * X[i,b]
//   X = in_values: (4096, 8192) fp32   (d_in[0])
//   W = weights  : (4096, 4096) fp32   (d_in[1])
//   out          : (4096, 8192) fp32
//
// Compute-bound fp32 GEMM as bf16x3 split (hi*hi + lo*hi + hi*lo) on
// v_wmma_f32_16x16x32_bf16 with fp32 accumulation.
// Pipeline per k-step (BLK 128x256x32, 8 waves):
//   async global->LDS fp32 staging (double-buffered, ASYNCcnt)
//   -> producer phase: each element of the block tile split ONCE into packed
//      bf16 hi/lo planes laid out in WMMA pair order (48 elts/lane)
//   -> consumer phase: fragments via pure ds_load_b128, 48 WMMA/lane.

#define IN_F   4096
#define OUT_F  4096
#define BATCHN 8192

#define BLK_M 128
#define BLK_N 256
#define BLK_K 32
#define A_STRIDE 132                   // fp32 staging row stride (pad, 16B align)
#define B_STRIDE 260
#define TILE_A_F (BLK_K * A_STRIDE)    // 4224 floats
#define TILE_B_F (BLK_K * B_STRIDE)    // 8320 floats
#define BUF_F (TILE_A_F + TILE_B_F)    // 12544 floats per staging buffer

#define P_STRIDE 20                    // bf16-plane row stride in u32 (16 used)
#define AH_OFF (2 * BUF_F)             // u32 offsets into shmem
#define AL_OFF (AH_OFF + BLK_M * P_STRIDE)
#define BH_OFF (AL_OFF + BLK_M * P_STRIDE)
#define BL_OFF (BH_OFF + BLK_N * P_STRIDE)
#define SH_U32 (BL_OFF + BLK_N * P_STRIDE)   // 40448 u32 = 161,792 B

typedef __attribute__((ext_vector_type(16))) __bf16   v16bf;
typedef __attribute__((ext_vector_type(2)))  __bf16   v2bf;
typedef __attribute__((ext_vector_type(8)))  float    v8f;
typedef __attribute__((ext_vector_type(8)))  unsigned v8u;
typedef __attribute__((ext_vector_type(4)))  unsigned v4u;
typedef int v4i_t __attribute__((vector_size(16)));  // matches builtin param pointee

#if defined(__AMDGCN__) && __has_builtin(__builtin_amdgcn_global_load_async_to_lds_b128)
#define HAS_ASYNC_LDS 1
#else
#define HAS_ASYNC_LDS 0
#endif

#if HAS_ASYNC_LDS
#if __has_builtin(__builtin_amdgcn_s_wait_asynccnt)
#define WAIT_ASYNC(n) __builtin_amdgcn_s_wait_asynccnt(n)
#else
#define WAIT_ASYNC(n) asm volatile("s_wait_asynccnt %0" ::"i"(n) : "memory")
#endif
#else
#define WAIT_ASYNC(n)
#endif

__device__ __forceinline__ void copy_f4(const float* g, float* l) {
#if HAS_ASYNC_LDS
  __builtin_amdgcn_global_load_async_to_lds_b128((v4i_t*)g, (v4i_t*)l,
                                                 /*offset=*/0, /*cpol=*/0);
#else
  *(float4*)l = *(const float4*)g;
#endif
}

// Copy one 32x128 W tile and one 32x256 X tile into fp32 staging `buf`.
// 256 threads: 4 float4 for A + 8 float4 for B per thread = 12 async ops/wave.
__device__ __forceinline__ void load_tile(float* buf, const float* __restrict__ W,
                                          const float* __restrict__ X, int K0, int M0,
                                          int N0, int tid) {
  const int ccolA = (tid & 31) * 4;  // 0..124
  const int crowA = tid >> 5;        // 0..7
#pragma unroll
  for (int r = 0; r < 4; ++r) {
    const int k = crowA + r * 8;
    copy_f4(&W[(size_t)(K0 + k) * OUT_F + M0 + ccolA], &buf[k * A_STRIDE + ccolA]);
  }
  const int ccolB = (tid & 63) * 4;  // 0..252
  const int crowB = tid >> 6;        // 0..3
#pragma unroll
  for (int r = 0; r < 8; ++r) {
    const int k = crowB + r * 4;
    copy_f4(&X[(size_t)(K0 + k) * BATCHN + N0 + ccolB],
            &buf[TILE_A_F + k * B_STRIDE + ccolB]);
  }
}

// Split (f0,f1) -> packed bf16 hi pair (bit truncation, one v_perm_b32) and
// packed bf16 lo residual pair.
__device__ __forceinline__ void split_pair(float f0, float f1, unsigned& hpair,
                                           unsigned& lpair) {
  const unsigned u0 = __builtin_bit_cast(unsigned, f0);
  const unsigned u1 = __builtin_bit_cast(unsigned, f1);
  hpair = __builtin_amdgcn_perm(u1, u0, 0x07060302u);  // [f1.hi16 : f0.hi16]
  const v2bf hp = __builtin_bit_cast(v2bf, hpair);
  v2bf lp;
  lp[0] = (__bf16)(f0 - (float)hp[0]);
  lp[1] = (__bf16)(f1 - (float)hp[1]);
  lpair = __builtin_bit_cast(unsigned, lp);
}

// Convert 8 staged fp32 (one k-octet of one row) into one hi and one lo uint4.
__device__ __forceinline__ void convert_octet(const float* col, int stride,
                                              v4u& hv, v4u& lv) {
#pragma unroll
  for (int j = 0; j < 4; ++j) {
    const float f0 = col[(2 * j) * stride];
    const float f1 = col[(2 * j + 1) * stride];
    unsigned hp, lp;
    split_pair(f0, f1, hp, lp);
    hv[j] = hp;
    lv[j] = lp;
  }
}

// Build a 16-bf16 fragment from two 16B plane reads (ds_load_b128 x2).
__device__ __forceinline__ v16bf frag_from(const unsigned* p0, const unsigned* p1) {
  const v4u a = *(const v4u*)p0;
  const v4u b = *(const v4u*)p1;
  v8u w;
  w[0] = a[0]; w[1] = a[1]; w[2] = a[2]; w[3] = a[3];
  w[4] = b[0]; w[5] = b[1]; w[6] = b[2]; w[7] = b[3];
  return __builtin_bit_cast(v16bf, w);
}

__global__ __launch_bounds__(256) void sparse_mm_bf16x3_kernel(
    const float* __restrict__ X, const float* __restrict__ W, float* __restrict__ out) {
  __shared__ __align__(16) unsigned shmem[SH_U32];
  float*    stg = (float*)shmem;         // [2][A 32x132 | B 32x260] fp32
  unsigned* AHP = shmem + AH_OFF;        // [128][20] u32: A hi pairs, [m][k/2]
  unsigned* ALP = shmem + AL_OFF;        // [128][20] u32: A lo pairs
  unsigned* BHP = shmem + BH_OFF;        // [256][20] u32: B hi pairs, [n][k/2]
  unsigned* BLP = shmem + BL_OFF;        // [256][20] u32: B lo pairs

  const int tid  = threadIdx.x;
  const int lane = tid & 31;
  const int wid  = tid >> 5;   // 0..7
  const int wm   = wid & 1;    // wave M half    (64 rows of 128)
  const int wn   = wid >> 1;   // wave N quarter (64 cols of 256)
  const int ln   = lane & 15;  // row (A) / col (B,C) within 16x16 tile
  const int half = lane >> 4;

  const int M0 = blockIdx.y * BLK_M;
  const int N0 = blockIdx.x * BLK_N;

  v8f acc[4][4] = {};

  const int NK = IN_F / BLK_K;  // 128 k-steps

  load_tile(&stg[0], W, X, 0, M0, N0, tid);

  for (int ks = 0; ks < NK; ++ks) {
    if (ks + 1 < NK) {
      load_tile(&stg[((ks + 1) & 1) * BUF_F], W, X, (ks + 1) * BLK_K, M0, N0, tid);
      WAIT_ASYNC(12);  // 12 outstanding = the just-issued prefetch batch
    } else {
      WAIT_ASYNC(0);
    }
    __syncthreads();  // staging ready; previous consume phase done

    const float* SA = &stg[(ks & 1) * BUF_F];  // [k][m], 32 x 128 (+pad)
    const float* SB = SA + TILE_A_F;           // [k][n], 32 x 256 (+pad)

    // ---- producer: split each block element once into bf16 hi/lo planes ----
    // A: 128 rows x 4 k-octets = 512 tasks; 2 per thread.
#pragma unroll
    for (int r = 0; r < 2; ++r) {
      const int task = tid + 256 * r;
      const int m = task >> 2;
      const int oct = task & 3;        // k0 = oct*8
      v4u hv, lv;
      convert_octet(&SA[(oct * 8) * A_STRIDE + m], A_STRIDE, hv, lv);
      *(v4u*)&AHP[m * P_STRIDE + oct * 4] = hv;
      *(v4u*)&ALP[m * P_STRIDE + oct * 4] = lv;
    }
    // B: 256 rows x 4 k-octets = 1024 tasks; 4 per thread.
#pragma unroll
    for (int r = 0; r < 4; ++r) {
      const int task = tid + 256 * r;
      const int n = task >> 2;
      const int oct = task & 3;
      v4u hv, lv;
      convert_octet(&SB[(oct * 8) * B_STRIDE + n], B_STRIDE, hv, lv);
      *(v4u*)&BHP[n * P_STRIDE + oct * 4] = hv;
      *(v4u*)&BLP[n * P_STRIDE + oct * 4] = lv;
    }

    __syncthreads();  // planes ready

    // ---- consumer: fragments via ds_load_b128, then 48 WMMA ----
    // A 16x32 layout: VGPRs 0-3 <- k pairs half*4+0..3, VGPRs 4-7 <- 8+half*4+0..3
    v16bf ah[4], al[4];
#pragma unroll
    for (int mf = 0; mf < 4; ++mf) {
      const int m = wm * 64 + mf * 16 + ln;
      ah[mf] = frag_from(&AHP[m * P_STRIDE + half * 4], &AHP[m * P_STRIDE + 8 + half * 4]);
      al[mf] = frag_from(&ALP[m * P_STRIDE + half * 4], &ALP[m * P_STRIDE + 8 + half * 4]);
    }
#pragma unroll
    for (int nf = 0; nf < 4; ++nf) {
      // B 32x16 layout: VGPRs 0-3 <- k pairs half*4+0..3, VGPRs 4-7 <- 8+half*4+0..3
      const int n = wn * 64 + nf * 16 + ln;
      const v16bf bh =
          frag_from(&BHP[n * P_STRIDE + half * 4], &BHP[n * P_STRIDE + 8 + half * 4]);
      const v16bf bl =
          frag_from(&BLP[n * P_STRIDE + half * 4], &BLP[n * P_STRIDE + 8 + half * 4]);
#pragma unroll
      for (int mf = 0; mf < 4; ++mf) {
        acc[mf][nf] = __builtin_amdgcn_wmma_f32_16x16x32_bf16(
            false, ah[mf], false, bh, (short)0, acc[mf][nf], false, false);
        acc[mf][nf] = __builtin_amdgcn_wmma_f32_16x16x32_bf16(
            false, al[mf], false, bh, (short)0, acc[mf][nf], false, false);
        acc[mf][nf] = __builtin_amdgcn_wmma_f32_16x16x32_bf16(
            false, ah[mf], false, bl, (short)0, acc[mf][nf], false, false);
      }
    }
    // next iteration's first barrier separates this consume from next produce
  }

  // ---- epilogue: C/D layout lane=N, VGPR v -> M = half*8 + v ----
#pragma unroll
  for (int mf = 0; mf < 4; ++mf) {
#pragma unroll
    for (int nf = 0; nf < 4; ++nf) {
      const size_t n = (size_t)N0 + wn * 64 + nf * 16 + ln;
#pragma unroll
      for (int v = 0; v < 8; ++v) {
        const size_t m = (size_t)M0 + wm * 64 + mf * 16 + half * 8 + v;
        out[m * BATCHN + n] = acc[mf][nf][v];
      }
    }
  }
}

extern "C" void kernel_launch(void* const* d_in, const int* in_sizes, int n_in,
                              void* d_out, int out_size, void* d_ws, size_t ws_size,
                              hipStream_t stream) {
  (void)in_sizes; (void)n_in; (void)out_size; (void)d_ws; (void)ws_size;
  const float* X = (const float*)d_in[0];  // in_values (4096, 8192)
  const float* W = (const float*)d_in[1];  // weights   (4096, 4096)
  float* out = (float*)d_out;              // (4096, 8192)

  dim3 grid(BATCHN / BLK_N, OUT_F / BLK_M);  // (32, 32)
  dim3 block(256);
  hipLaunchKernelGGL(sparse_mm_bf16x3_kernel, grid, block, 0, stream, X, W, out);
}
